// Net_16922171146622
// MI455X (gfx1250) — matrix operand order repaired and checked by reading, not compiled, not run on hardware
//
#include <hip/hip_runtime.h>
#include <hip/hip_bf16.h>

typedef __attribute__((ext_vector_type(16))) _Float16 v16h;
typedef __attribute__((ext_vector_type(8)))  float    v8f;

#define NB   4
#define NN   4096
#define NK   20
#define NP   (NB * NN)        // 16384 points
#define NE   (NP * NK)        // 327680 edges
#define EPSF 1e-5f
#define SLP  0.2f

// ---- ordered-uint mapping so we can use integer atomicMax for float max ----
__device__ __forceinline__ unsigned fkey(float f) {
  unsigned u = __float_as_uint(f);
  return (u & 0x80000000u) ? ~u : (u | 0x80000000u);
}
__device__ __forceinline__ float funkey(unsigned u) {
  return __uint_as_float((u & 0x80000000u) ? (u & 0x7fffffffu) : ~u);
}

// =====================================================================
// Kernel 1: kNN (top-20 by -squared-distance), one wave32 per query point
// =====================================================================
__global__ __launch_bounds__(256) void k_knn(const float* __restrict__ x,
                                             int* __restrict__ idxbuf) {
  __shared__ float candd[8][640];
  __shared__ int   candi[8][640];
  const int wv   = threadIdx.x >> 5;
  const int lane = threadIdx.x & 31;
  const int pt   = blockIdx.x * 8 + wv;       // 0..16383
  const int b    = pt >> 12;
  const int n    = pt & 4095;
  const float* xb = x + b * 3 * NN;
  const float qx = xb[n], qy = xb[NN + n], qz = xb[2 * NN + n];

  float bd[NK]; int bi[NK];
#pragma unroll
  for (int t = 0; t < NK; ++t) { bd[t] = -__builtin_inff(); bi[t] = 0; }

  for (int j = lane; j < NN; j += 32) {
    __builtin_prefetch(&xb[j + 256], 0, 3);   // global_prefetch_b8
    float dx = xb[j] - qx, dy = xb[NN + j] - qy, dz = xb[2 * NN + j] - qz;
    float d = -(dx * dx + dy * dy + dz * dz);   // pd: larger == nearer
    if (d > bd[NK - 1]) {
      bd[NK - 1] = d; bi[NK - 1] = j;
#pragma unroll
      for (int t = NK - 1; t >= 1; --t) {       // one bubble pass keeps list sorted
        if (bd[t] > bd[t - 1]) {
          float tv = bd[t]; bd[t] = bd[t - 1]; bd[t - 1] = tv;
          int   ti = bi[t]; bi[t] = bi[t - 1]; bi[t - 1] = ti;
        }
      }
    }
  }
#pragma unroll
  for (int t = 0; t < NK; ++t) {
    candd[wv][lane * NK + t] = bd[t];
    candi[wv][lane * NK + t] = bi[t];
  }
  __syncthreads();
  for (int r = 0; r < NK; ++r) {
    float mv = -__builtin_inff(); int mp = 0;
#pragma unroll
    for (int t = 0; t < 20; ++t) {              // 640 / 32 lanes
      int pos = lane + t * 32;
      float v = candd[wv][pos];
      if (v > mv) { mv = v; mp = pos; }
    }
#pragma unroll
    for (int s = 16; s >= 1; s >>= 1) {         // wave32 butterfly argmax
      float ov = __shfl_xor(mv, s);
      int   op = __shfl_xor(mp, s);
      if (ov > mv) { mv = ov; mp = op; }
    }
    if (lane == 0) {
      idxbuf[pt * NK + r] = candi[wv][mp];
      candd[wv][mp] = -__builtin_inff();
    }
    __syncthreads();
  }
}

// =====================================================================
// Kernel 2: swizzle W1 (384x64 f32) into WMMA A-fragment layout, f16.
// Layout: [24 row-tiles][2 k-frags][32 lanes][16 halves], 32B/lane contig.
// Contiguous K order (matches B): lane<16 holds K {0..15},
// lane>=16 holds K {16..31} (plus kfrag*32), row M = tile*16 + lane%16.
// =====================================================================
__global__ void k_w1swz(const float* __restrict__ w1, _Float16* __restrict__ w1sw) {
  int t = blockIdx.x * 256 + threadIdx.x;
  if (t >= 24 * 2 * 32 * 16) return;
  int p    = t & 15;
  int lane = (t >> 4) & 31;
  int kf   = (t >> 9) & 1;
  int rt   = t >> 10;
  int M  = rt * 16 + (lane & 15);
  int hs = (lane >> 4) * 8;
  int K  = kf * 32 + hs * 2 + p;
  w1sw[t] = (_Float16)w1[M * 64 + K];
}

// =====================================================================
// Kernel 3: init x1 (ordered-uint encoded) to -inf
// =====================================================================
__global__ void k_x1init(unsigned* __restrict__ x1u) {
  int i = blockIdx.x * 256 + threadIdx.x;
  if (i < NP * 64) x1u[i] = 0x007FFFFFu;   // fkey(-inf)
}

// =====================================================================
// Kernel 4: fused AGConv. One block = 16 edges.
//  - gather feat(6) from x + idx
//  - h = lrelu(bn0(W0 feat)) -> f16 LDS (B-fragment friendly [edge][64])
//  - y tile = W1 @ h via v_wmma_f32_16x16x32_f16 (M=chan, N=edge, K=64)
//  - z[e][o] = sum_c y[6o+c][e] * feat[e][c]; bn1+lrelu; k-max via atomicMax
// =====================================================================
__global__ __launch_bounds__(256) void k_agconv(
    const float* __restrict__ x, const int* __restrict__ idxbuf,
    const float* __restrict__ w0, const float* __restrict__ g0, const float* __restrict__ b0,
    const v16h* __restrict__ w1swv,
    const float* __restrict__ g1, const float* __restrict__ b1,
    unsigned* __restrict__ x1u) {
  __shared__ float featLds[16][8];
  __shared__ __align__(32) _Float16 hLds[16][64];
  __shared__ float w0Lds[384];
  __shared__ float s0[64], be0[64], s1[64], be1[64];

  const int t    = threadIdx.x;
  const int w    = t >> 5;
  const int lane = t & 31;

  // stage weights / bn params
  w0Lds[t] = w0[t];
  if (t < 128) w0Lds[256 + t] = w0[256 + t];
  if (t < 64) {
    float r = rsqrtf(1.f + EPSF);
    s0[t] = g0[t] * r;  be0[t] = b0[t];
    s1[t] = g1[t] * r;  be1[t] = b1[t];
  }
  // gather edge features
  if (t < 16) {
    int e = blockIdx.x * 16 + t;
    int p = e / NK;
    int b = p >> 12, n = p & 4095;
    int j = idxbuf[e];
    const float* xb = x + b * 3 * NN;
    float cx = xb[n], cy = xb[NN + n], cz = xb[2 * NN + n];
    float px = xb[j], py = xb[NN + j], pz = xb[2 * NN + j];
    featLds[t][0] = px - cx; featLds[t][1] = py - cy; featLds[t][2] = pz - cz;
    featLds[t][3] = cx;      featLds[t][4] = cy;      featLds[t][5] = cz;
  }
  __syncthreads();

  // h = lrelu(bn0(W0 @ feat)): 16 edges x 64 ch, 4 ch per thread
  {
    int e = t & 15, mg = t >> 4;
#pragma unroll
    for (int q = 0; q < 4; ++q) {
      int m = mg * 4 + q;
      float acc = 0.f;
#pragma unroll
      for (int c = 0; c < 6; ++c) acc += w0Lds[m * 6 + c] * featLds[e][c];
      acc = acc * s0[m] + be0[m];
      acc = (acc >= 0.f) ? acc : SLP * acc;
      hLds[e][m] = (_Float16)acc;
    }
  }
  __syncthreads();

  // B fragment (32x16 f16): lane<16 -> edge=lane, K 0..15; lane>=16 -> K 16..31
  const int el    = lane & 15;
  const int khalf = (lane >> 4) * 16;
  v16h bf0 = *reinterpret_cast<const v16h*>(&hLds[el][khalf]);
  v16h bf1 = *reinterpret_cast<const v16h*>(&hLds[el][32 + khalf]);

  v8f acc[3];
#pragma unroll
  for (int rt = 0; rt < 3; ++rt) {              // wave w: channels 48w..48w+47
    v16h a0 = w1swv[((w * 3 + rt) * 2 + 0) * 32 + lane];
    v16h a1 = w1swv[((w * 3 + rt) * 2 + 1) * 32 + lane];
    v8f c = {};
    c = __builtin_amdgcn_wmma_f32_16x16x32_f16(false, a0, false, bf0, (short)0, c, false, false);
    c = __builtin_amdgcn_wmma_f32_16x16x32_f16(false, a1, false, bf1, (short)0, c, false, false);
    acc[rt] = c;
  }

  // contraction z[e][o] = sum_c y[6o+c][e]*feat[e][c], o in [8w, 8w+8).
  // Branch on lane half so rel (and thus rel/6, rel%6) are compile-time
  // constants -> part[] stays in registers with static indices (no cndmask
  // emulation of dynamic register indexing).
  float feat6[6];
#pragma unroll
  for (int c = 0; c < 6; ++c) feat6[c] = featLds[el][c];

  float part[8];
#pragma unroll
  for (int o = 0; o < 8; ++o) part[o] = 0.f;

  if (lane < 16) {                              // D rows M = 0..7 of each tile
#pragma unroll
    for (int rt = 0; rt < 3; ++rt) {
#pragma unroll
      for (int r = 0; r < 8; ++r) {
        const int rel = rt * 16 + r;            // compile-time constant
        part[rel / 6] += acc[rt][r] * feat6[rel % 6];
      }
    }
  } else {                                      // D rows M = 8..15 of each tile
#pragma unroll
    for (int rt = 0; rt < 3; ++rt) {
#pragma unroll
      for (int r = 0; r < 8; ++r) {
        const int rel = rt * 16 + r + 8;        // compile-time constant
        part[rel / 6] += acc[rt][r] * feat6[rel % 6];
      }
    }
  }

  int e = blockIdx.x * 16 + el;
  int p = e / NK;
#pragma unroll
  for (int o = 0; o < 8; ++o) {
    float z = part[o] + __shfl_xor(part[o], 16); // combine row halves (lane ^ 16)
    int og = w * 8 + o;
    z = z * s1[og] + be1[og];
    z = (z >= 0.f) ? z : SLP * z;
    if (lane < 16) atomicMax(&x1u[p * 64 + og], fkey(z));
  }
}

// =====================================================================
// Kernel 5: out = lrelu(bn_out(w_out @ x1)), one thread per point
// =====================================================================
__global__ void k_final(const unsigned* __restrict__ x1u,
                        const float* __restrict__ w_out,
                        const float* __restrict__ g_out,
                        const float* __restrict__ b_out,
                        float* __restrict__ out) {
  int p = blockIdx.x * 256 + threadIdx.x;
  if (p >= NP) return;
  int b = p >> 12, n = p & 4095;
  float a0 = 0.f, a1 = 0.f, a2 = 0.f;
#pragma unroll 8
  for (int c = 0; c < 64; ++c) {
    float v = funkey(x1u[p * 64 + c]);
    a0 += w_out[c] * v;
    a1 += w_out[64 + c] * v;
    a2 += w_out[128 + c] * v;
  }
  float r = rsqrtf(1.f + EPSF);
  float o0 = a0 * (g_out[0] * r) + b_out[0];
  float o1 = a1 * (g_out[1] * r) + b_out[1];
  float o2 = a2 * (g_out[2] * r) + b_out[2];
  o0 = (o0 >= 0.f) ? o0 : SLP * o0;
  o1 = (o1 >= 0.f) ? o1 : SLP * o1;
  o2 = (o2 >= 0.f) ? o2 : SLP * o2;
  out[(b * 3 + 0) * NN + n] = o0;
  out[(b * 3 + 1) * NN + n] = o1;
  out[(b * 3 + 2) * NN + n] = o2;
}

extern "C" void kernel_launch(void* const* d_in, const int* in_sizes, int n_in,
                              void* d_out, int out_size, void* d_ws, size_t ws_size,
                              hipStream_t stream) {
  (void)in_sizes; (void)n_in; (void)out_size; (void)ws_size;
  const float* x     = (const float*)d_in[0];
  const float* w0    = (const float*)d_in[1];
  const float* g0    = (const float*)d_in[2];
  const float* b0    = (const float*)d_in[3];
  const float* w1    = (const float*)d_in[4];
  const float* g1    = (const float*)d_in[5];
  const float* b1    = (const float*)d_in[6];
  const float* w_out = (const float*)d_in[7];
  const float* g_out = (const float*)d_in[8];
  const float* b_out = (const float*)d_in[9];
  float* out = (float*)d_out;

  // workspace layout (all offsets 256B aligned): ~5.6 MB total
  char* ws = (char*)d_ws;
  int*       idxbuf = (int*)ws;                                  // NE*4      = 1,310,720 B
  _Float16*  w1sw   = (_Float16*)(ws + 1310720);                 // 24576*2   =    49,152 B
  unsigned*  x1u    = (unsigned*)(ws + 1310720 + 49152);         // NP*64*4   = 4,194,304 B

  k_knn   <<<NP / 8, 256, 0, stream>>>(x, idxbuf);
  k_w1swz <<<96, 256, 0, stream>>>(w1, w1sw);
  k_x1init<<<(NP * 64) / 256, 256, 0, stream>>>(x1u);
  k_agconv<<<NE / 16, 256, 0, stream>>>(x, idxbuf, w0, g0, b0,
                                        (const v16h*)w1sw, g1, b1, x1u);
  k_final <<<NP / 256, 256, 0, stream>>>(x1u, w_out, g_out, b_out, out);
}
